// SparseConvNeXtBlock3d_14568529068655
// MI455X (gfx1250) — compile-verified, matrix operand order checked by reading, not run on hardware
//
#include <hip/hip_runtime.h>
#include <cstdint>

#define N_VOXELS 500000
#define C_DIM    128
#define HID_DIM  512
#define K_OFFS   27
#define TILE_M   128
#define LDA      136   // padded u16 stride for bf16 LDS tiles (272B rows -> bank spread)
#define LDH      132   // padded f32 stride for conv-result scratch
#define TILE_BYTES (TILE_M * LDA * 2)   // 34816

typedef __attribute__((ext_vector_type(16))) __bf16   v16bf;
typedef __attribute__((ext_vector_type(8)))  float    v8f;
typedef __attribute__((ext_vector_type(8)))  uint16_t v8u16;

struct __align__(16) BfFragBits { v8u16 lo, hi; };

static __device__ __forceinline__ uint16_t f2bf(float f) {
  uint32_t u = __float_as_uint(f);
  u += 0x7FFFu + ((u >> 16) & 1u);       // round-to-nearest-even
  return (uint16_t)(u >> 16);
}

static __device__ __forceinline__ v16bf load_frag(const uint16_t* base) {
  BfFragBits b;
  b.lo = *(const v8u16*)(base);
  b.hi = *(const v8u16*)(base + 16);
  return __builtin_bit_cast(v16bf, b);
}

static __device__ __forceinline__ v8f zero8() {
  v8f z = {0.f, 0.f, 0.f, 0.f, 0.f, 0.f, 0.f, 0.f};
  return z;
}

// ---- CDNA5 async global->LDS copy (ASYNCcnt-tracked, no VGPR data movement) ----
// Copies 128 bytes: 8 x b128, one instruction offset advances both LDS and
// global addresses (ISA 15.18.3: dsaddr = VDST + INST_OFFSET, mem = VADDR + INST_OFFSET).
static __device__ __forceinline__ void async_copy128(uint32_t ldsOff, const uint16_t* src) {
  uint64_t ga = (uint64_t)(uintptr_t)src;
#pragma unroll
  for (int i = 0; i < 8; ++i)
    asm volatile("global_load_async_to_lds_b128 %0, %1, off offset:%2"
                 :: "v"(ldsOff), "v"(ga), "i"(i * 16) : "memory");
}

static __device__ __forceinline__ void wait_async0() {
  asm volatile("s_wait_asynccnt 0x0" ::: "memory");
}

// ---------------- prep kernels: fp32 -> bf16 (with transposes) ----------------

__global__ __launch_bounds__(256) void prep_weights(
    const float* __restrict__ conv_w, const float* __restrict__ w1,
    const float* __restrict__ w2, uint16_t* __restrict__ wconv_t,
    uint16_t* __restrict__ w1t, uint16_t* __restrict__ w2t) {
  const int NCONV = K_OFFS * C_DIM * C_DIM;            // 442368
  const int NW1   = C_DIM * HID_DIM;                   // 65536
  int tid = blockIdx.x * 256 + threadIdx.x;
  if (tid < NCONV) {
    int k = tid / (C_DIM * C_DIM);
    int rem = tid - k * (C_DIM * C_DIM);
    int co = rem >> 7, ci = rem & 127;                 // dst [k][co][ci]
    wconv_t[tid] = f2bf(conv_w[k * C_DIM * C_DIM + ci * C_DIM + co]);
  } else if (tid < NCONV + NW1) {
    int t = tid - NCONV;
    int hid = t >> 7, c = t & 127;                     // w1t[hid][c]
    w1t[t] = f2bf(w1[c * HID_DIM + hid]);
  } else if (tid < NCONV + 2 * NW1) {
    int t = tid - NCONV - NW1;
    int c = t / HID_DIM, hid = t - c * HID_DIM;        // w2t[c][hid]
    w2t[t] = f2bf(w2[hid * C_DIM + c]);
  }
}

__global__ __launch_bounds__(256) void prep_feats(
    const float* __restrict__ feats, uint16_t* __restrict__ f16) {
  int tid = blockIdx.x * 256 + threadIdx.x;
  const int total = (N_VOXELS + 1) * C_DIM;
  if (tid < total) {
    int row = tid >> 7;
    f16[tid] = (row < N_VOXELS) ? f2bf(feats[tid]) : (uint16_t)0;  // zero pad row
  }
}

// ---------------- fused conv27 + LN + MLP + residual ----------------

__global__ __launch_bounds__(256) void fused_block(
    const float* __restrict__ feats, const int* __restrict__ nbr,
    const float* __restrict__ conv_b, const float* __restrict__ ln_g,
    const float* __restrict__ ln_b, const float* __restrict__ b1,
    const float* __restrict__ b2, const uint16_t* __restrict__ feats16,
    const uint16_t* __restrict__ wconv_t, const uint16_t* __restrict__ w1t,
    const uint16_t* __restrict__ w2t, float* __restrict__ out) {
  // 4 ping-pong bf16 tiles + 2KB reduce scratch; fp32 LN scratch aliases T2+T3.
  __shared__ __align__(16) unsigned char smem[4 * TILE_BYTES + 2048];
  uint16_t* T0   = (uint16_t*)(smem);                   // conv A ping0 -> hn
  uint16_t* T1   = (uint16_t*)(smem + 1 * TILE_BYTES);  // conv B ping0 -> w1 buf
  uint16_t* T2   = (uint16_t*)(smem + 2 * TILE_BYTES);  // conv A ping1 -> sH lo / hidden
  uint16_t* T3   = (uint16_t*)(smem + 3 * TILE_BYTES);  // conv B ping1 -> sH hi / w2 buf
  float*    sH   = (float*)(smem + 2 * TILE_BYTES);     // 128x132 f32 (67584B <= 2 tiles)
  float*    sRed = (float*)(smem + 4 * TILE_BYTES);     // 512 f32

  const int tid  = threadIdx.x;
  const int lane = tid & 31;
  const int wv   = tid >> 5;
  const int l15  = lane & 15;
  const int h    = lane >> 4;
  const int mOff = h * 8;          // C/D: lanes16-31 hold rows M=r+8
  const int mBase = wv * 16;       // wave's 16-row strip
  const int tileBase = blockIdx.x * TILE_M;
  const int cr = tid >> 1;         // cooperative-copy row (0..127)
  const int ch = (tid & 1) * 64;   // half-row offset in u16

  // issue async gather(row k) + weight tile(k) into a ping-pong pair
  auto issue_conv = [&](int k, uint16_t* At, uint16_t* Bt) {
    int row = tileBase + cr;
    int gidx = N_VOXELS;           // zero row
    if (row < N_VOXELS) {
      int t = nbr[row * K_OFFS + k];
      gidx = (t < 0) ? N_VOXELS : t;
    }
    async_copy128((uint32_t)(uintptr_t)(At + cr * LDA + ch),
                  feats16 + (size_t)gidx * C_DIM + ch);
    async_copy128((uint32_t)(uintptr_t)(Bt + cr * LDA + ch),
                  wconv_t + (size_t)k * C_DIM * C_DIM + cr * C_DIM + ch);
  };

  // 128x128x128 bf16 GEMM step; depth-2 software pipeline on B fragments so the
  // pre-WMMA dscnt wait only covers the previous load pair while the next flies.
  auto wmma_block = [&](const uint16_t* At, const uint16_t* Bt, v8f* accArr) {
    const uint16_t* arow = At + (mBase + l15) * LDA + 8 * h;
    const uint16_t* brow = Bt + l15 * LDA + 8 * h;
#pragma unroll
    for (int kt = 0; kt < 4; ++kt) {
      v16bf afrag = load_frag(arow + kt * 32);
      v16bf bcur = load_frag(brow + kt * 32);
#pragma unroll
      for (int nt = 0; nt < 8; ++nt) {
        v16bf bnxt = bcur;
        if (nt < 7) bnxt = load_frag(brow + (nt + 1) * 16 * LDA + kt * 32);
        accArr[nt] = __builtin_amdgcn_wmma_f32_16x16x32_bf16(
            false, afrag, false, bcur, (short)0, accArr[nt], false, false);
        bcur = bnxt;
      }
    }
  };

  // ---------------- sparse conv: 27 x (async gather + bf16 GEMM), LDS ping-pong ----
  v8f acc[8];
#pragma unroll
  for (int i = 0; i < 8; ++i) acc[i] = zero8();

  issue_conv(0, T0, T1);
  for (int k = 0; k < K_OFFS - 1; k += 2) {
    wait_async0();
    __syncthreads();
    issue_conv(k + 1, T2, T3);     // loads fly while we compute on T0/T1
    wmma_block(T0, T1, acc);

    wait_async0();
    __syncthreads();
    if (k + 2 < K_OFFS) issue_conv(k + 2, T0, T1);
    wmma_block(T2, T3, acc);
  }
  // k = 26 (even -> T0/T1)
  wait_async0();
  __syncthreads();
  wmma_block(T0, T1, acc);

  // conv result + bias -> sH (fp32); T2/T3 readers retired at last barrier
#pragma unroll
  for (int nt = 0; nt < 8; ++nt) {
    int n = nt * 16 + l15;
    float bn = conv_b[n];
#pragma unroll
    for (int r = 0; r < 8; ++r)
      sH[(mBase + mOff + r) * LDH + n] = acc[nt][r] + bn;
  }
  __syncthreads();

  // prefetch w1 chunk0 into T1 (overlaps with LayerNorm)
  async_copy128((uint32_t)(uintptr_t)(T1 + cr * LDA + ch),
                w1t + (size_t)cr * C_DIM + ch);

  // ---------------- LayerNorm (fp32) -> hn bf16 into T0 (2 threads per row) ----
  {
    const int r    = tid >> 1;
    const int half = tid & 1;
    const float* hrow = sH + r * LDH + half * 64;
    float s = 0.f, ss = 0.f;
#pragma unroll 4
    for (int j = 0; j < 64; ++j) { float v = hrow[j]; s += v; ss += v * v; }
    sRed[tid]       = s;
    sRed[256 + tid] = ss;
    __syncthreads();
    float stot  = sRed[2 * r] + sRed[2 * r + 1];
    float sstot = sRed[256 + 2 * r] + sRed[256 + 2 * r + 1];
    float mu   = stot * (1.f / C_DIM);
    float var  = sstot * (1.f / C_DIM) - mu * mu;
    float rstd = rsqrtf(var + 1e-6f);
    uint16_t* drow = T0 + r * LDA + half * 64;
    const float* g = ln_g + half * 64;
    const float* b = ln_b + half * 64;
#pragma unroll 4
    for (int j = 0; j < 64; ++j)
      drow[j] = f2bf((hrow[j] - mu) * rstd * g[j] + b[j]);
  }
  wait_async0();    // w1 chunk0 landed in T1
  __syncthreads();

  // ---------------- MLP: hn @ w1 -> SiLU -> @ w2, per 128-col hidden chunk -----
  // T0 = hn (A), T1 = w1 chunk buf, T2 = hidden (sC), T3 = w2 chunk buf
  v8f outAcc[8];
#pragma unroll
  for (int i = 0; i < 8; ++i) outAcc[i] = zero8();

  for (int chunk = 0; chunk < 4; ++chunk) {
    // w2 chunk load overlaps GEMM1
    async_copy128((uint32_t)(uintptr_t)(T3 + cr * LDA + ch),
                  w2t + (size_t)cr * HID_DIM + chunk * 128 + ch);

    v8f hid[8];
#pragma unroll
    for (int i = 0; i < 8; ++i) hid[i] = zero8();
    wmma_block(T0, T1, hid);                      // GEMM1: hn @ w1[chunk]

    // SiLU(+b1) -> T2 (bf16); GEMM2 reads only this wave's rows (DS in-order)
#pragma unroll
    for (int nt = 0; nt < 8; ++nt) {
      int n = nt * 16 + l15;
      float bv = b1[chunk * 128 + n];
#pragma unroll
      for (int r = 0; r < 8; ++r) {
        float x = hid[nt][r] + bv;
        float sg = 1.f / (1.f + __expf(-x));
        T2[(mBase + mOff + r) * LDA + n] = f2bf(x * sg);
      }
    }
    wait_async0();      // w2 chunk in T3
    __syncthreads();    // also retires all GEMM1 readers of T1

    if (chunk + 1 < 4)  // next w1 chunk load overlaps GEMM2
      async_copy128((uint32_t)(uintptr_t)(T1 + cr * LDA + ch),
                    w1t + (size_t)((chunk + 1) * 128 + cr) * C_DIM + ch);

    wmma_block(T2, T3, outAcc);                   // GEMM2: silu @ w2[chunk]

    wait_async0();      // w1 chunk+1 in T1
    __syncthreads();    // retires GEMM2 readers of T3 before next overwrite
  }

  // ---------------- epilogue: + b2 + residual(feats), fp32 store ----------------
#pragma unroll
  for (int nt = 0; nt < 8; ++nt) {
    int n = nt * 16 + l15;
    float bv = b2[n];
#pragma unroll
    for (int r = 0; r < 8; ++r) {
      int row = tileBase + mBase + mOff + r;
      if (row < N_VOXELS) {
        size_t o = (size_t)row * C_DIM + n;
        out[o] = outAcc[nt][r] + bv + feats[o];
      }
    }
  }
}

// ---------------- launcher ----------------

extern "C" void kernel_launch(void* const* d_in, const int* in_sizes, int n_in,
                              void* d_out, int out_size, void* d_ws, size_t ws_size,
                              hipStream_t stream) {
  const float* feats  = (const float*)d_in[0];
  const int*   nbr    = (const int*)d_in[1];
  const float* conv_w = (const float*)d_in[2];
  const float* conv_b = (const float*)d_in[3];
  const float* ln_g   = (const float*)d_in[4];
  const float* ln_b   = (const float*)d_in[5];
  const float* w1     = (const float*)d_in[6];
  const float* b1     = (const float*)d_in[7];
  const float* w2     = (const float*)d_in[8];
  const float* b2     = (const float*)d_in[9];
  float* out = (float*)d_out;
  (void)in_sizes; (void)n_in; (void)out_size; (void)ws_size;

  uint8_t* ws = (uint8_t*)d_ws;
  // byte offsets, 256B aligned
  const size_t OFF_FEATS16 = 0;                         // (500001*128*2)=128000256 -> pad 128000512
  const size_t OFF_WCONV   = 128000512;                 // 27*128*128*2 = 884736
  const size_t OFF_W1T     = OFF_WCONV + 884736;        // 512*128*2 = 131072
  const size_t OFF_W2T     = OFF_W1T + 131072;          // 128*512*2 = 131072
  uint16_t* feats16 = (uint16_t*)(ws + OFF_FEATS16);
  uint16_t* wconv_t = (uint16_t*)(ws + OFF_WCONV);
  uint16_t* w1t     = (uint16_t*)(ws + OFF_W1T);
  uint16_t* w2t     = (uint16_t*)(ws + OFF_W2T);

  {
    const int total = K_OFFS * C_DIM * C_DIM + 2 * C_DIM * HID_DIM;  // 573440
    prep_weights<<<(total + 255) / 256, 256, 0, stream>>>(conv_w, w1, w2, wconv_t, w1t, w2t);
  }
  {
    const int total = (N_VOXELS + 1) * C_DIM;                        // 64000128
    prep_feats<<<(total + 255) / 256, 256, 0, stream>>>(feats, feats16);
  }
  {
    const int tiles = (N_VOXELS + TILE_M - 1) / TILE_M;              // 3907
    fused_block<<<tiles, 256, 0, stream>>>(feats, nbr, conv_b, ln_g, ln_b, b1, b2,
                                           feats16, wconv_t, w1t, w2t, out);
  }
}
